// GraphSageNet_55207509623126
// MI455X (gfx1250) — matrix operand order repaired and checked by reading, not compiled, not run on hardware
//
#include <hip/hip_runtime.h>
#include <hip/hip_bf16.h>
#include <math.h>

#define HID 108
#define NT  7      // 7 column tiles of 16 -> 112 (108 padded)
#define BM  128    // rows per block: 8 waves x 16

typedef __attribute__((ext_vector_type(16))) _Float16 v16h;
typedef __attribute__((ext_vector_type(8)))  float    v8f;

// ---------------------------------------------------------------- utilities
__global__ void zero_f(float* __restrict__ p, long n) {
  long i = (long)blockIdx.x * blockDim.x + threadIdx.x;
  if (i < n) p[i] = 0.0f;
}

__global__ void degree_k(const int* __restrict__ dst, float* __restrict__ deg, int E) {
  int e = blockIdx.x * blockDim.x + threadIdx.x;
  if (e < E) atomicAdd(&deg[dst[e]], 1.0f);
}

__global__ void count_k(const int* __restrict__ gid, float* __restrict__ cnt, int N) {
  int i = blockIdx.x * blockDim.x + threadIdx.x;
  if (i < N) atomicAdd(&cnt[gid[i]], 1.0f);
}

// ---------------------------------------------------------------- WMMA GEMM
// out[M x 108] = epilogue( X0[M x K0] * W0[K0 x 108]
//                        (+ X1[M x K1] * W1[K1 x 108]) + bias )
// f16 inputs staged in LDS in exact A/B fragment layout, f32 accumulate.
__launch_bounds__(256, 2)
__global__ void gemm_wmma(const float* __restrict__ X0, const float* __restrict__ W0,
                          const float* __restrict__ X1, const float* __restrict__ W1,
                          const float* __restrict__ bias, float* __restrict__ out,
                          int M, int K0, int K1, int relu)
{
  // B fragments: sW[tile][kchunk][lane*16+i] ; A fragments: sX[wave][kchunk][lane*16+i]
  __shared__ __align__(32) _Float16 sW[NT][4][512];  // 28 KB
  __shared__ __align__(32) _Float16 sX[8][4][512];   // 32 KB
  const int tid  = threadIdx.x;
  const int wave = tid >> 5;
  const int lane = tid & 31;
  const int row0 = blockIdx.x * BM;

  v8f acc[NT] = {};

  for (int p = 0; p < 2; ++p) {
    const float* X = p ? X1 : X0;
    const float* W = p ? W1 : W0;
    const int K = p ? K1 : K0;
    if (K == 0) break;
    const int nkc = (K + 31) >> 5;

    __syncthreads();  // LDS reuse across passes
    // Stage weights, pre-swizzled to B-matrix 32x16 f16 layout:
    //   column n = lane&15, k = (i&7) | ((i&8)<<1) | ((lane&16)>>1)  (ISA 7.12.2)
    for (int idx = tid; idx < NT * nkc * 512; idx += 256) {
      int i    = idx & 15;
      int ln   = (idx >> 4) & 31;
      int rest = idx >> 9;
      int kc   = rest % nkc;
      int t    = rest / nkc;
      int n = t * 16 + (ln & 15);
      int k = kc * 32 + (i & 7) + ((i & 8) << 1) + ((ln & 16) >> 1);
      float v = (k < K && n < HID) ? W[(size_t)k * HID + n] : 0.0f;
      sW[t][kc][ln * 16 + i] = (_Float16)v;
    }
    // Stage activations, pre-swizzled to A-matrix 16x32 f16 layout:
    //   row m = lane&15, k = (i&7) | ((i&8)<<1) | ((lane&16)>>1)
    for (int idx = tid; idx < 8 * nkc * 512; idx += 256) {
      int i    = idx & 15;
      int ln   = (idx >> 4) & 31;
      int rest = idx >> 9;
      int kc   = rest % nkc;
      int w    = rest / nkc;
      int m = row0 + w * 16 + (ln & 15);
      int k = kc * 32 + (i & 7) + ((i & 8) << 1) + ((ln & 16) >> 1);
      float v = (m < M && k < K) ? X[(size_t)m * K + k] : 0.0f;
      sX[w][kc][ln * 16 + i] = (_Float16)v;
    }
    __syncthreads();

    for (int kc = 0; kc < nkc; ++kc) {
      v16h a = *(const v16h*)&sX[wave][kc][lane * 16];
      #pragma unroll
      for (int t = 0; t < NT; ++t) {
        v16h b = *(const v16h*)&sW[t][kc][lane * 16];
        acc[t] = __builtin_amdgcn_wmma_f32_16x16x32_f16(
            false, a, false, b, (short)0, acc[t], false, false);
      }
    }
  }

  // Epilogue: C/D layout m = r + 8*(lane>>4), n = lane&15
  const int mBase = row0 + wave * 16 + ((lane >> 4) << 3);
  const int nBase = lane & 15;
  #pragma unroll
  for (int t = 0; t < NT; ++t) {
    int n = t * 16 + nBase;
    if (n >= HID) continue;
    float bv = bias[n];
    #pragma unroll
    for (int r = 0; r < 8; ++r) {
      int m = mBase + r;
      if (m < M) {
        float v = acc[t][r] + bv;
        if (relu) v = fmaxf(v, 0.0f);
        out[(size_t)m * HID + n] = v;
      }
    }
  }
}

// ---------------------------------------------------------------- graph ops
// wave-per-edge: agg[dst] += hpool[src]  (108 floats, L2-resident atomics)
__global__ void scatter_add_k(const float* __restrict__ hpool, const int* __restrict__ src,
                              const int* __restrict__ dst, float* __restrict__ agg, int E) {
  int wid = (int)(((long)blockIdx.x * blockDim.x + threadIdx.x) >> 5);
  if (wid >= E) return;
  int lane = threadIdx.x & 31;
  const float* r = hpool + (size_t)src[wid] * HID;
  float*       o = agg   + (size_t)dst[wid] * HID;
  #pragma unroll
  for (int j = 0; j < 4; ++j) {
    int c = lane + 32 * j;
    if (c < HID) atomicAdd(&o[c], r[c]);
  }
}

__global__ void scale_invdeg_k(float* __restrict__ agg, const float* __restrict__ deg, long total) {
  long i = (long)blockIdx.x * blockDim.x + threadIdx.x;
  if (i < total) {
    int n = (int)(i / HID);
    agg[i] *= 1.0f / fmaxf(deg[n], 1.0f);
  }
}

// wave-per-row: h = h + relu(bundle / max(||bundle||2, eps))
__global__ void norm_relu_res_k(const float* __restrict__ bundle, float* __restrict__ h, int M) {
  int row = (int)(((long)blockIdx.x * blockDim.x + threadIdx.x) >> 5);
  if (row >= M) return;
  int lane = threadIdx.x & 31;
  const float* b  = bundle + (size_t)row * HID;
  float*       hp = h      + (size_t)row * HID;
  float v[4]; float ss = 0.0f;
  #pragma unroll
  for (int j = 0; j < 4; ++j) {
    int c = lane + 32 * j;
    v[j] = (c < HID) ? b[c] : 0.0f;
    ss += v[j] * v[j];
  }
  #pragma unroll
  for (int o = 16; o > 0; o >>= 1) ss += __shfl_xor(ss, o, 32);  // wave32 reduce
  float inv = 1.0f / fmaxf(sqrtf(ss), 1e-12f);
  #pragma unroll
  for (int j = 0; j < 4; ++j) {
    int c = lane + 32 * j;
    if (c < HID) hp[c] += fmaxf(v[j] * inv, 0.0f);
  }
}

__global__ void readout_sum_k(const float* __restrict__ h, const int* __restrict__ gid,
                              float* __restrict__ hg, int N) {
  int row = (int)(((long)blockIdx.x * blockDim.x + threadIdx.x) >> 5);
  if (row >= N) return;
  int lane = threadIdx.x & 31;
  int g = gid[row];
  #pragma unroll
  for (int j = 0; j < 4; ++j) {
    int c = lane + 32 * j;
    if (c < HID) atomicAdd(&hg[(size_t)g * HID + c], h[(size_t)row * HID + c]);
  }
}

// block-per-graph tiny MLP: 108 -> 54 -> 27 -> 10
__global__ void mlp_k(const float* __restrict__ hg, const float* __restrict__ cnt,
                      const float* __restrict__ W1, const float* __restrict__ b1,
                      const float* __restrict__ W2, const float* __restrict__ b2,
                      const float* __restrict__ W3, const float* __restrict__ b3,
                      float* __restrict__ out) {
  __shared__ float s0[HID];
  __shared__ float s1[54];
  __shared__ float s2[27];
  int g = blockIdx.x;
  float ic = 1.0f / fmaxf(cnt[g], 1.0f);
  for (int c = threadIdx.x; c < HID; c += blockDim.x) s0[c] = hg[(size_t)g * HID + c] * ic;
  __syncthreads();
  for (int o = threadIdx.x; o < 54; o += blockDim.x) {
    float a = b1[o];
    for (int k = 0; k < HID; ++k) a += s0[k] * W1[k * 54 + o];
    s1[o] = fmaxf(a, 0.0f);
  }
  __syncthreads();
  for (int o = threadIdx.x; o < 27; o += blockDim.x) {
    float a = b2[o];
    for (int k = 0; k < 54; ++k) a += s1[k] * W2[k * 27 + o];
    s2[o] = fmaxf(a, 0.0f);
  }
  __syncthreads();
  for (int o = threadIdx.x; o < 10; o += blockDim.x) {
    float a = b3[o];
    for (int k = 0; k < 27; ++k) a += s2[k] * W3[k * 10 + o];
    out[(size_t)g * 10 + o] = a;
  }
}

// ---------------------------------------------------------------- launcher
extern "C" void kernel_launch(void* const* d_in, const int* in_sizes, int n_in,
                              void* d_out, int out_size, void* d_ws, size_t ws_size,
                              hipStream_t stream) {
  (void)n_in; (void)ws_size;
  const float* nodes_feat = (const float*)d_in[0];
  const int*   src  = (const int*)d_in[4];
  const int*   dst  = (const int*)d_in[5];
  const int*   gid  = (const int*)d_in[6];
  const float* W_emb  = (const float*)d_in[7];
  const float* b_emb  = (const float*)d_in[8];
  const float* W_pool = (const float*)d_in[9];
  const float* b_pool = (const float*)d_in[10];
  const float* W_app  = (const float*)d_in[11];
  const float* b_app  = (const float*)d_in[12];
  const float* W1 = (const float*)d_in[13];
  const float* b1 = (const float*)d_in[14];
  const float* W2 = (const float*)d_in[15];
  const float* b2 = (const float*)d_in[16];
  const float* W3 = (const float*)d_in[17];
  const float* b3 = (const float*)d_in[18];

  const int N  = in_sizes[6];       // 50000
  const int E  = in_sizes[4];       // 800000
  const int Gn = out_size / 10;     // 256

  float* ws   = (float*)d_ws;
  size_t nh   = (size_t)N * HID;
  float* h    = ws;                 // [N,108]
  float* pool = ws + nh;            // [N,108]  h_pool, reused for bundle
  float* agg  = ws + 2 * nh;        // [N,108]
  float* deg  = ws + 3 * nh;        // [N]
  float* cnt  = deg + N;            // [G]
  float* hg   = cnt + Gn;           // [G,108]
  float* outp = (float*)d_out;

  auto cdiv = [](long a, long b) { return (int)((a + b - 1) / b); };

  // degrees / graph counts / readout accumulator (zero every call)
  long zlen = (long)N + Gn + (long)Gn * HID;
  zero_f<<<cdiv(zlen, 256), 256, 0, stream>>>(deg, zlen);
  degree_k<<<cdiv(E, 256), 256, 0, stream>>>(dst, deg, E);
  count_k<<<cdiv(N, 256), 256, 0, stream>>>(gid, cnt, N);

  const int gb = cdiv(N, BM);
  // input embedding: h = nodes_feat @ W_emb + b_emb
  gemm_wmma<<<gb, 256, 0, stream>>>(nodes_feat, W_emb, nullptr, nullptr,
                                    b_emb, h, N, 32, 0, /*relu=*/0);

  for (int l = 0; l < 4; ++l) {
    // h_pool = relu(h @ W_pool + b_pool)
    gemm_wmma<<<gb, 256, 0, stream>>>(h, W_pool + (size_t)l * HID * HID, nullptr, nullptr,
                                      b_pool + (size_t)l * HID, pool, N, HID, 0, /*relu=*/1);
    // agg = segment_sum(h_pool[src], dst) * inv_deg
    zero_f<<<cdiv((long)nh, 256), 256, 0, stream>>>(agg, (long)nh);
    scatter_add_k<<<cdiv(E, 8), 256, 0, stream>>>(pool, src, dst, agg, E);
    scale_invdeg_k<<<cdiv((long)nh, 256), 256, 0, stream>>>(agg, deg, (long)nh);
    // bundle = [h, agg] @ W_app + b_app   (fused as two K=108 accumulation passes)
    const float* Wt = W_app + (size_t)l * 2 * HID * HID;
    gemm_wmma<<<gb, 256, 0, stream>>>(h, Wt, agg, Wt + (size_t)HID * HID,
                                      b_app + (size_t)l * HID, pool, N, HID, HID, /*relu=*/0);
    // h = h + relu(l2norm(bundle))
    norm_relu_res_k<<<cdiv(N, 8), 256, 0, stream>>>(pool, h, N);
  }

  // mean_nodes readout + MLP head
  readout_sum_k<<<cdiv(N, 8), 256, 0, stream>>>(h, gid, hg, N);
  mlp_k<<<Gn, 128, 0, stream>>>(hg, cnt, W1, b1, W2, b2, W3, b3, outp);
}
